// ServiceGNN_59253368815670
// MI455X (gfx1250) — matrix operand order repaired and checked by reading, not compiled, not run on hardware
//
#include <hip/hip_runtime.h>

typedef __bf16 v16bf __attribute__((ext_vector_type(16)));
typedef float  v8f   __attribute__((ext_vector_type(8)));

#define DH   128
#define DOUT 57
#define DOUTP 64   // padded FC cols
#define NEG_SLOPE 0.2f

__device__ __forceinline__ __bf16 f2bf(float f) { return (__bf16)f; }

// ordered-uint encoding so atomicMax(uint) == float max
__device__ __forceinline__ unsigned fenc(float x) {
    unsigned u = __float_as_uint(x);
    return (u >> 31) ? ~u : (u | 0x80000000u);
}
__device__ __forceinline__ float fdec(unsigned u) {
    return (u & 0x80000000u) ? __uint_as_float(u & 0x7fffffffu) : __uint_as_float(~u);
}

// ---------------- one-time weight prep: Wt[col][k] = bf16(W[k][col]), 128x128
__global__ __launch_bounds__(256) void k_convW(const float* __restrict__ W,
                                               __bf16* __restrict__ Wt) {
    int idx = blockIdx.x * blockDim.x + threadIdx.x;
    if (idx >= DH * DH) return;
    int col = idx & (DH - 1);
    int k   = idx >> 7;
    Wt[(size_t)col * DH + k] = f2bf(W[(size_t)k * DH + col]);
}

// fc_w [128,57] -> Wtfc bf16 [64][128], zero-padded cols 57..63
__global__ __launch_bounds__(256) void k_convWfc(const float* __restrict__ W,
                                                 __bf16* __restrict__ Wt) {
    int idx = blockIdx.x * blockDim.x + threadIdx.x;
    if (idx >= DOUTP * DH) return;
    int col = idx & (DOUTP - 1);
    int k   = idx >> 6;
    float v = (col < DOUT) ? W[(size_t)k * DOUT + col] : 0.f;
    Wt[(size_t)col * DH + k] = f2bf(v);
}

// ---------------- GEMM: H[nrows,128] = X[nrows,128] @ W (Wt bf16 [col][k]) via WMMA
__global__ __launch_bounds__(256) void k_gemm128(const float* __restrict__ X,
                                                 const __bf16* __restrict__ Wt,
                                                 float* __restrict__ H, int nrows) {
    const int wave = threadIdx.x >> 5;   // 8 waves: one 16-col tile each
    const int lane = threadIdx.x & 31;
    const int half = lane >> 4;
    const int l16  = lane & 15;
    const int row0 = blockIdx.x * 16;
    if (row0 >= nrows) return;
    const int col0 = wave * 16;

    v8f acc = {};
    #pragma unroll
    for (int k0 = 0; k0 < DH; k0 += 32) {
        // A: 16x32 bf16. lane<16: K = k0+0..15 ; lane>=16: K = k0+16..31
        v16bf a;
        {
            const float* ap = X + (size_t)(row0 + l16) * DH;
            const int b0 = k0 + half * 16;
            const int b1 = b0 + 8;
            float4 x0 = *(const float4*)(ap + b0);
            float4 x1 = *(const float4*)(ap + b0 + 4);
            float4 x2 = *(const float4*)(ap + b1);
            float4 x3 = *(const float4*)(ap + b1 + 4);
            a[0]=f2bf(x0.x); a[1]=f2bf(x0.y); a[2]=f2bf(x0.z); a[3]=f2bf(x0.w);
            a[4]=f2bf(x1.x); a[5]=f2bf(x1.y); a[6]=f2bf(x1.z); a[7]=f2bf(x1.w);
            a[8]=f2bf(x2.x); a[9]=f2bf(x2.y); a[10]=f2bf(x2.z); a[11]=f2bf(x2.w);
            a[12]=f2bf(x3.x); a[13]=f2bf(x3.y); a[14]=f2bf(x3.z); a[15]=f2bf(x3.w);
        }
        // B: 32x16 bf16, pre-transposed: contiguous 16 bf16 per lane
        v16bf b = *(const v16bf*)(Wt + (size_t)(col0 + l16) * DH + k0 + half * 16);
        acc = __builtin_amdgcn_wmma_f32_16x16x32_bf16(false, a, false, b,
                                                      (short)0, acc, false, false);
    }
    float* hp = H + (size_t)row0 * DH + col0 + l16;
    #pragma unroll
    for (int r = 0; r < 8; ++r) hp[(size_t)(r + half * 8) * DH] = acc[r];
}

// ---------------- alpha_s/alpha_d: wave-per-node dot products
__global__ __launch_bounds__(256) void k_alpha(const float* __restrict__ H,
                                               const float* __restrict__ a_src,
                                               const float* __restrict__ a_dst,
                                               float* __restrict__ as_, float* __restrict__ ad_,
                                               int n) {
    const int wid  = (int)((blockIdx.x * (size_t)blockDim.x + threadIdx.x) >> 5);
    const int lane = threadIdx.x & 31;
    if (wid >= n) return;
    float4 hv = *(const float4*)(H + (size_t)wid * DH + lane * 4);
    float4 s  = *(const float4*)(a_src + lane * 4);
    float4 d  = *(const float4*)(a_dst + lane * 4);
    float ps = hv.x * s.x + hv.y * s.y + hv.z * s.z + hv.w * s.w;
    float pd = hv.x * d.x + hv.y * d.y + hv.z * d.z + hv.w * d.w;
    #pragma unroll
    for (int off = 16; off > 0; off >>= 1) {
        ps += __shfl_xor(ps, off, 32);
        pd += __shfl_xor(pd, off, 32);
    }
    if (lane == 0) { as_[wid] = ps; ad_[wid] = pd; }
}

// ---------------- zero agg, reset segment max/den
__global__ __launch_bounds__(256) void k_init(float* __restrict__ agg,
                                              unsigned* __restrict__ menc,
                                              float* __restrict__ den, int n, int total) {
    int i = blockIdx.x * blockDim.x + threadIdx.x;
    if (i < n) { menc[i] = 0u; den[i] = 0.f; }
    if (i < total) agg[i] = 0.f;
}

// ---------------- pass 1: per-edge logit + segment max(dst)
__global__ __launch_bounds__(256) void k_edge_logit(const long long* __restrict__ ei,
                                                    const float* __restrict__ as_,
                                                    const float* __restrict__ ad_,
                                                    float* __restrict__ logit,
                                                    unsigned* __restrict__ menc,
                                                    int E_, int n) {
    int i = blockIdx.x * blockDim.x + threadIdx.x;
    int tot = E_ + n;
    if (i >= tot) return;
    int s, d;
    if (i < E_) { s = (int)ei[i]; d = (int)ei[E_ + i]; }
    else        { s = i - E_;     d = s; }
    float e = as_[s] + ad_[d];
    e = (e > 0.f) ? e : NEG_SLOPE * e;
    logit[i] = e;
    atomicMax(menc + d, fenc(e));
}

// ---------------- pass 2: exp(e - m[dst]) + segment sum(dst)
__global__ __launch_bounds__(256) void k_edge_exp(const long long* __restrict__ ei,
                                                  float* __restrict__ logit,
                                                  const unsigned* __restrict__ menc,
                                                  float* __restrict__ den,
                                                  int E_, int n) {
    int i = blockIdx.x * blockDim.x + threadIdx.x;
    int tot = E_ + n;
    if (i >= tot) return;
    int d = (i < E_) ? (int)ei[E_ + i] : (i - E_);
    float v = __expf(logit[i] - fdec(menc[d]));
    logit[i] = v;
    atomicAdd(den + d, v);
}

// ---------------- pass 3: wave-per-edge weighted scatter agg[dst] += w*h[src]
__global__ __launch_bounds__(256) void k_edge_agg(const long long* __restrict__ ei,
                                                  const float* __restrict__ H,
                                                  const float* __restrict__ logit,
                                                  const float* __restrict__ den,
                                                  float* __restrict__ agg,
                                                  int E_, int n) {
    const int widx = (int)((blockIdx.x * (size_t)blockDim.x + threadIdx.x) >> 5);
    const int lane = threadIdx.x & 31;
    int tot = E_ + n;
    if (widx >= tot) return;
    int s, d;
    if (widx < E_) { s = (int)ei[widx]; d = (int)ei[E_ + widx]; }
    else           { s = widx - E_;     d = s; }
    float w = logit[widx] / den[d];
    float4 hv = *(const float4*)(H + (size_t)s * DH + lane * 4);
    float* ap = agg + (size_t)d * DH + lane * 4;
    atomicAdd(ap + 0, w * hv.x);
    atomicAdd(ap + 1, w * hv.y);
    atomicAdd(ap + 2, w * hv.z);
    atomicAdd(ap + 3, w * hv.w);
}

// ---------------- elementwise bias (+optional relu)
__global__ __launch_bounds__(256) void k_bias_relu(float* __restrict__ feat,
                                                   const float* __restrict__ b,
                                                   int total, int do_relu) {
    int i = blockIdx.x * blockDim.x + threadIdx.x;
    if (i >= total) return;
    float v = feat[i] + b[i & (DH - 1)];
    feat[i] = (do_relu && v < 0.f) ? 0.f : v;
}

// ---------------- FC: out[n,57] = (A + b2) @ fc_w + fc_b (Wtfc bf16 [64][128], padded)
__global__ __launch_bounds__(128) void k_fc(const float* __restrict__ A,
                                            const float* __restrict__ b2,
                                            const __bf16* __restrict__ Wtfc,
                                            const float* __restrict__ bf,
                                            float* __restrict__ out, int nrows) {
    const int wave = threadIdx.x >> 5;   // 4 waves => cols 0..63 (guard >=57 on store)
    const int lane = threadIdx.x & 31;
    const int half = lane >> 4;
    const int l16  = lane & 15;
    const int row0 = blockIdx.x * 16;
    if (row0 >= nrows) return;
    const int col  = wave * 16 + l16;

    v8f acc = {};
    #pragma unroll
    for (int k0 = 0; k0 < DH; k0 += 32) {
        v16bf a;
        {
            const float* ap = A + (size_t)(row0 + l16) * DH;
            const int b0 = k0 + half * 16;
            const int b1 = b0 + 8;
            #pragma unroll
            for (int j = 0; j < 8; ++j) a[j]     = f2bf(ap[b0 + j] + b2[b0 + j]);
            #pragma unroll
            for (int j = 0; j < 8; ++j) a[8 + j] = f2bf(ap[b1 + j] + b2[b1 + j]);
        }
        v16bf b = *(const v16bf*)(Wtfc + (size_t)col * DH + k0 + half * 16);
        acc = __builtin_amdgcn_wmma_f32_16x16x32_bf16(false, a, false, b,
                                                      (short)0, acc, false, false);
    }
    if (col < DOUT) {
        float bb = bf[col];
        #pragma unroll
        for (int r = 0; r < 8; ++r)
            out[(size_t)(row0 + r + half * 8) * DOUT + col] = acc[r] + bb;
    }
}

extern "C" void kernel_launch(void* const* d_in, const int* in_sizes, int n_in,
                              void* d_out, int out_size, void* d_ws, size_t ws_size,
                              hipStream_t stream) {
    const float*     x      = (const float*)d_in[0];
    const long long* ei     = (const long long*)d_in[1];
    const float*     W1     = (const float*)d_in[2];
    const float*     a1_src = (const float*)d_in[3];
    const float*     a1_dst = (const float*)d_in[4];
    const float*     b1     = (const float*)d_in[5];
    const float*     W2     = (const float*)d_in[6];
    const float*     a2_src = (const float*)d_in[7];
    const float*     a2_dst = (const float*)d_in[8];
    const float*     b2     = (const float*)d_in[9];
    const float*     fc_w   = (const float*)d_in[10];
    const float*     fc_b   = (const float*)d_in[11];
    float*           out    = (float*)d_out;

    const int N_  = in_sizes[0] / DH;   // 50000
    const int E_  = in_sizes[1] / 2;    // 800000
    const int TOT = E_ + N_;            // edges incl. self-loops

    // workspace layout
    float*    h    = (float*)d_ws;                     // N*128 f32
    float*    feat = h + (size_t)N_ * DH;              // N*128 f32 (also agg)
    float*    as_  = feat + (size_t)N_ * DH;           // N
    float*    ad_  = as_ + N_;                         // N
    unsigned* menc = (unsigned*)(ad_ + N_);            // N
    float*    den  = (float*)(menc + N_);              // N
    float*    exb  = den + N_;                         // E+N
    __bf16*   wt1  = (__bf16*)(exb + TOT);             // 128*128 bf16
    __bf16*   wt2  = wt1 + DH * DH;                    // 128*128 bf16
    __bf16*   wtfc = wt2 + DH * DH;                    // 64*128 bf16

    const int B = 256;
    const int gRows  = (N_ + 15) / 16;
    const int gWaveN = (N_ * 32 + B - 1) / B;
    const int gElem  = (N_ * DH + B - 1) / B;
    const int gEdge  = (TOT + B - 1) / B;
    const int gWaveE = (int)(((size_t)TOT * 32 + B - 1) / B);

    // ---- one-time weight prep (bf16, transposed, WMMA-ready) ----
    k_convW<<<(DH * DH + B - 1) / B, B, 0, stream>>>(W1, wt1);
    k_convW<<<(DH * DH + B - 1) / B, B, 0, stream>>>(W2, wt2);
    k_convWfc<<<(DOUTP * DH + B - 1) / B, B, 0, stream>>>(fc_w, wtfc);

    // ---- layer 1 ----
    k_gemm128<<<gRows, 256, 0, stream>>>(x, wt1, h, N_);
    k_alpha<<<gWaveN, B, 0, stream>>>(h, a1_src, a1_dst, as_, ad_, N_);
    k_init<<<gElem, B, 0, stream>>>(feat, menc, den, N_, N_ * DH);      // feat <- agg
    k_edge_logit<<<gEdge, B, 0, stream>>>(ei, as_, ad_, exb, menc, E_, N_);
    k_edge_exp<<<gEdge, B, 0, stream>>>(ei, exb, menc, den, E_, N_);
    k_edge_agg<<<gWaveE, B, 0, stream>>>(ei, h, exb, den, feat, E_, N_);
    k_bias_relu<<<gElem, B, 0, stream>>>(feat, b1, N_ * DH, 1);

    // ---- layer 2 ----
    k_gemm128<<<gRows, 256, 0, stream>>>(feat, wt2, h, N_);
    k_alpha<<<gWaveN, B, 0, stream>>>(h, a2_src, a2_dst, as_, ad_, N_);
    k_init<<<gElem, B, 0, stream>>>(feat, menc, den, N_, N_ * DH);      // feat <- agg2
    k_edge_logit<<<gEdge, B, 0, stream>>>(ei, as_, ad_, exb, menc, E_, N_);
    k_edge_exp<<<gEdge, B, 0, stream>>>(ei, exb, menc, den, E_, N_);
    k_edge_agg<<<gWaveE, B, 0, stream>>>(ei, h, exb, den, feat, E_, N_);

    // ---- FC: out = (agg2 + b2) @ fc_w + fc_b ----
    k_fc<<<gRows, 128, 0, stream>>>(feat, b2, wtfc, fc_b, out, N_);
}
